// RSRLinear_44513041056293
// MI455X (gfx1250) — 8-bit matrix operand order repaired and checked by reading, compiled, not run on hardware
//
#include <hip/hip_runtime.h>
#include <stdint.h>

#define IN_F   4096
#define OUT_F  4096
#define NTOK   8192
#define RMS_EPS 1e-6f
#define Q_EPS   1e-5f

typedef int v8i __attribute__((ext_vector_type(8)));

__global__ void __launch_bounds__(256)
pack_w_kernel(const float* __restrict__ W, int8_t* __restrict__ Wq) {
    const int idx = (blockIdx.x * 256 + threadIdx.x) * 4;
    float4 v = *(const float4*)(W + idx);
    char4 c;
    c.x = (char)(int)v.x;
    c.y = (char)(int)v.y;
    c.z = (char)(int)v.z;
    c.w = (char)(int)v.w;
    *(char4*)(Wq + idx) = c;
}

__global__ void __launch_bounds__(256)
rmsq_kernel(const float* __restrict__ x, const float* __restrict__ rw,
            int8_t* __restrict__ q, float* __restrict__ dq) {
    __shared__ float red[256];
    const int tok = blockIdx.x;
    const int tid = threadIdx.x;
    const float* xr = x + (size_t)tok * IN_F + tid * 16;
    const float* wr = rw + tid * 16;

    float4 xv[4], wv[4];
    float ss = 0.f;
#pragma unroll
    for (int i = 0; i < 4; ++i) {
        xv[i] = *(const float4*)(xr + i * 4);
        wv[i] = *(const float4*)(wr + i * 4);
        ss += xv[i].x * xv[i].x + xv[i].y * xv[i].y +
              xv[i].z * xv[i].z + xv[i].w * xv[i].w;
    }
    red[tid] = ss;
    __syncthreads();
    for (int s = 128; s > 0; s >>= 1) {
        if (tid < s) red[tid] += red[tid + s];
        __syncthreads();
    }
    const float sumsq = red[0];
    __syncthreads();
    const float r = rsqrtf(sumsq * (1.0f / (float)IN_F) + RMS_EPS);

    float h[16];
    float amax = 0.f;
#pragma unroll
    for (int i = 0; i < 4; ++i) {
        h[i * 4 + 0] = xv[i].x * r * wv[i].x;
        h[i * 4 + 1] = xv[i].y * r * wv[i].y;
        h[i * 4 + 2] = xv[i].z * r * wv[i].z;
        h[i * 4 + 3] = xv[i].w * r * wv[i].w;
#pragma unroll
        for (int j = 0; j < 4; ++j) amax = fmaxf(amax, fabsf(h[i * 4 + j]));
    }
    red[tid] = amax;
    __syncthreads();
    for (int s = 128; s > 0; s >>= 1) {
        if (tid < s) red[tid] = fmaxf(red[tid], red[tid + s]);
        __syncthreads();
    }
    const float am    = fmaxf(red[0], Q_EPS);
    const float scale = 127.0f / am;
    if (tid == 0) dq[tok] = am * (1.0f / 127.0f);

    union { int8_t b[16]; int4 v; } u;
#pragma unroll
    for (int j = 0; j < 16; ++j) {
        int v = (int)rintf(h[j] * scale);
        v = v < -128 ? -128 : (v > 127 ? 127 : v);
        u.b[j] = (int8_t)v;
    }
    *(int4*)(q + (size_t)tok * IN_F + tid * 16) = u.v;
}

__global__ void __launch_bounds__(256)
gemm_kernel(const int8_t* __restrict__ q, const int8_t* __restrict__ wq,
            const float* __restrict__ dq, const float* __restrict__ bias,
            const float* __restrict__ wscale_p, float* __restrict__ out) {
    const int lane   = threadIdx.x & 31;
    const int wave   = threadIdx.x >> 5;
    const int m_tile = blockIdx.y * 8 + wave;
    const int m_base = m_tile * 16;
    const int n_base = blockIdx.x * 64;
    const int half   = lane >> 4;
    const int l15    = lane & 15;

    const int8_t* arow  = q  + (size_t)(m_base + l15) * IN_F + (half ? 16 : 0);
    const int8_t* brow0 = wq + (size_t)(n_base + l15) * IN_F + (half ? 16 : 0);

    v8i acc[4] = {};

    for (int k = 0; k < IN_F; k += 64) {
        int2 t0 = *(const int2*)(arow + k + 0);
        int2 t1 = *(const int2*)(arow + k + 8);
        int2 t2 = *(const int2*)(arow + k + 32);
        int2 t3 = *(const int2*)(arow + k + 40);
        v8i a;
        a[0] = t0.x; a[1] = t0.y; a[2] = t1.x; a[3] = t1.y;
        a[4] = t2.x; a[5] = t2.y; a[6] = t3.x; a[7] = t3.y;

        __builtin_prefetch(arow + k + 128, 0, 0);

#pragma unroll
        for (int t = 0; t < 4; ++t) {
            const int8_t* bp = brow0 + (size_t)t * 16 * IN_F + k;
            int4 u0 = *(const int4*)(bp);
            int4 u1 = *(const int4*)(bp + 32);
            v8i b;
            b[0] = u0.x; b[1] = u0.y; b[2] = u0.z; b[3] = u0.w;
            b[4] = u1.x; b[5] = u1.y; b[6] = u1.z; b[7] = u1.w;
            acc[t] = __builtin_amdgcn_wmma_i32_16x16x64_iu8(
                true, a, true, b, acc[t], false, false);
        }
    }

    const float ws = wscale_p[0];
#pragma unroll
    for (int t = 0; t < 4; ++t) {
        const int ncol = n_base + t * 16 + l15;
        const float bc = bias[ncol];
#pragma unroll
        for (int r = 0; r < 8; ++r) {
            const int m = m_base + (half ? 8 + r : r);
            const float y = ((float)acc[t][r] * dq[m] + bc) * ws;
            out[(size_t)m * OUT_F + ncol] = y;
        }
    }
}

extern "C" void kernel_launch(void* const* d_in, const int* in_sizes, int n_in,
                              void* d_out, int out_size, void* d_ws, size_t ws_size,
                              hipStream_t stream) {
    const float* x      = (const float*)d_in[0];
    const float* W      = (const float*)d_in[1];
    const float* rmsw   = (const float*)d_in[2];
    const float* bias   = (const float*)d_in[3];
    const float* wscale = (const float*)d_in[4];
    float* out = (float*)d_out;

    int8_t* Wq = (int8_t*)d_ws;
    int8_t* q  = Wq + (size_t)OUT_F * IN_F;
    float*  dq = (float*)(q + (size_t)NTOK * IN_F);

    pack_w_kernel<<<(OUT_F * IN_F) / (256 * 4), 256, 0, stream>>>(W, Wq);

    rmsq_kernel<<<NTOK, 256, 0, stream>>>(x, rmsw, q, dq);

    dim3 grid(OUT_F / 64, NTOK / (16 * 8));
    gemm_kernel<<<grid, 256, 0, stream>>>(q, Wq, dq, bias, wscale, out);
}
